// Mamba2_37383395344510
// MI455X (gfx1250) — compile-verified, matrix operand order checked by reading, not run on hardware
//
#include <hip/hip_runtime.h>

// ---------------------------------------------------------------------------
// Mamba2 block for MI455X (gfx1250, wave32, WMMA, Tensor Data Mover).
// All matmuls go through v_wmma_f32_16x16x32_bf16. GEMM tiles are DMA'd into
// LDS by the TDM (tensor_load_to_lds, TENSORcnt) with double buffering; the
// SSD scan is fused per (batch, head) with the running 64x128 state held in
// WMMA accumulator VGPRs across all 64 chunks.
// ---------------------------------------------------------------------------

typedef __attribute__((ext_vector_type(16))) __bf16  v16bf;
typedef __attribute__((ext_vector_type(8)))  float   v8f;
typedef uint32_t v4u __attribute__((ext_vector_type(4)));
typedef uint32_t v8u __attribute__((ext_vector_type(8)));

#define WMMA_BF16(A, B, C) \
  __builtin_amdgcn_wmma_f32_16x16x32_bf16(false, (A), false, (B), (short)0, (C), false, false)

// Model constants
#define DMODEL   2048
#define DINNER   4096
#define DSTATE   128
#define NHEADS   64
#define HEADDIM  64
#define CHUNK    64
#define BATCH    2
#define SEQLEN   4096
#define CONVDIM  4352          // DINNER + 2*DSTATE
#define DINPROJ  8512          // 2*DINNER + 2*DSTATE + NHEADS
#define MROWS    8192          // BATCH * SEQLEN

__device__ __forceinline__ v8f vzero8() {
  v8f z;
#pragma unroll
  for (int r = 0; r < 8; ++r) z[r] = 0.0f;
  return z;
}

// Low 32 bits of a generic pointer to LDS == LDS byte offset (flat aperture).
__device__ __forceinline__ uint32_t lds_off(const void* p) {
  return (uint32_t)(uintptr_t)p;
}

// ---------------------------------------------------------------------------
// Tensor Data Mover: 2-D bf16 tile (tile_w x tile_h elements) from global
// (row pitch row_stride elements) into LDS (rows packed contiguously).
// D# layout per CDNA5 ISA:
//   group0: [1:0]=count=1 | [63:32]=lds_addr | [120:64]=global_addr | type=2
//   group1: data_size=1 (2B), tensor_dim0/1, tile_dim0/1, tensor_dim0_stride
// Issued by one wave; tracked with TENSORcnt (s_wait_tensorcnt).
// ---------------------------------------------------------------------------
__device__ __forceinline__ void tdm_load_2d_bf16(uint32_t lds_addr, const void* gptr,
                                                 uint32_t tile_w, uint32_t tile_h,
                                                 uint32_t row_stride) {
  uint64_t ga = (uint64_t)gptr;
  uint32_t tw = row_stride;        // tensor_dim0: full row length (no OOB clip)
  uint32_t th = 1u << 20;          // tensor_dim1: large

  v4u g0;
  g0[0] = 1u;                                                  // count=1, user mode
  g0[1] = lds_addr;                                            // LDS byte address
  g0[2] = (uint32_t)ga;                                        // global_addr[31:0]
  g0[3] = (uint32_t)((ga >> 32) & 0x01FFFFFFu) | (2u << 30);   // addr[56:32] | type=2

  v8u g1;
  g1[0] = 1u << 16;                                            // data_size=2 bytes
  g1[1] = (tw & 0xFFFFu) << 16;                                // tensor_dim0[15:0]
  g1[2] = (tw >> 16) | ((th & 0xFFFFu) << 16);                 // td0[31:16] | td1[15:0]
  g1[3] = (th >> 16) | ((tile_w & 0xFFFFu) << 16);             // td1[31:16] | tile_dim0
  g1[4] = tile_h & 0xFFFFu;                                    // tile_dim1 (tile_dim2=0)
  g1[5] = row_stride;                                          // tensor_dim0_stride[31:0]
  g1[6] = 0u;                                                  // stride[47:32], td1_stride lo
  g1[7] = 0u;

  asm volatile("tensor_load_to_lds %0, %1" :: "s"(g0), "s"(g1) : "memory");
}

// Fragment load for a matrix stored row-major with rows along K.
__device__ __forceinline__ v16bf frag_row(const __bf16* base, int row0, int ld,
                                          int k0, int lane) {
  const __bf16* p = base + (size_t)(row0 + (lane & 15)) * ld + k0 + ((lane >> 4) << 4);
  return *(const v16bf*)p;
}

// B-operand fragment gathered from a [K][ldn] row-major tile (transpose gather).
__device__ __forceinline__ v16bf frag_gatherT(const __bf16* base, int n0, int ldn,
                                              int k0, int lane) {
  int n  = n0 + (lane & 15);
  int kb = k0 + ((lane >> 4) << 4);
  v16bf r;
#pragma unroll
  for (int i = 0; i < 16; ++i) r[i] = base[(kb + i) * ldn + n];
  return r;
}

// ---------------------------------------------------------------------------
// f32 -> bf16 convert
// ---------------------------------------------------------------------------
__global__ __launch_bounds__(256) void cvt_bf16(const float* __restrict__ in,
                                                __bf16* __restrict__ out, int n) {
  int i = blockIdx.x * 256 + threadIdx.x;
  if (i < n) out[i] = (__bf16)in[i];
}

// ---------------------------------------------------------------------------
// C[M,N] = A[M,K] * B[N,K]^T (bf16, both K-contiguous), f32 out.
// 256 threads = 8 waves (4Mx2N); block tile 128x64; wave tile 32x32 (2x2 WMMA).
// TDM double-buffered pipeline: wave 0 issues tensor_load_to_lds for tile k+32
// while all waves run WMMAs on tile k; one barrier per K step.
// ---------------------------------------------------------------------------
__global__ __launch_bounds__(256) void gemm_bf16_nt(const __bf16* __restrict__ A,
                                                    const __bf16* __restrict__ B,
                                                    float* __restrict__ C,
                                                    int M, int N, int K) {
  __shared__ __bf16 As[2][128 * 32];
  __shared__ __bf16 Bs[2][64 * 32];

  const int tid  = threadIdx.x;
  const int lane = tid & 31;
  const int wave = tid >> 5;
  const int wm   = wave >> 1;      // 0..3
  const int wn   = wave & 1;       // 0..1
  const int m0   = blockIdx.y * 128;
  const int n0   = blockIdx.x * 64;
  const bool issuer = (wave == 0);

  v8f acc[2][2];
#pragma unroll
  for (int i = 0; i < 2; ++i)
#pragma unroll
    for (int j = 0; j < 2; ++j) acc[i][j] = vzero8();

  const int cl = lane & 15;
  const int rb = (lane >> 4) << 3;

  // Preload tile kk=0 into buffer 0 via TDM.
  if (issuer) {
    tdm_load_2d_bf16(lds_off(&As[0][0]), A + (size_t)m0 * K, 32, 128, (uint32_t)K);
    tdm_load_2d_bf16(lds_off(&Bs[0][0]), B + (size_t)n0 * K, 32, 64,  (uint32_t)K);
    __builtin_amdgcn_s_wait_tensorcnt(0);
  }
  __syncthreads();

  int cur = 0;
  for (int kk = 0; kk < K; kk += 32) {
    bool more = (kk + 32) < K;
    if (issuer && more) {  // DMA next tile while we compute on this one
      tdm_load_2d_bf16(lds_off(&As[cur ^ 1][0]), A + (size_t)m0 * K + kk + 32, 32, 128,
                       (uint32_t)K);
      tdm_load_2d_bf16(lds_off(&Bs[cur ^ 1][0]), B + (size_t)n0 * K + kk + 32, 32, 64,
                       (uint32_t)K);
    }

    const __bf16* Ab = &As[cur][0];
    const __bf16* Bb = &Bs[cur][0];
    v16bf af0 = frag_row(Ab, wm * 32,      32, 0, lane);
    v16bf af1 = frag_row(Ab, wm * 32 + 16, 32, 0, lane);
    v16bf bf0 = frag_row(Bb, wn * 32,      32, 0, lane);
    v16bf bf1 = frag_row(Bb, wn * 32 + 16, 32, 0, lane);

    acc[0][0] = WMMA_BF16(af0, bf0, acc[0][0]);
    acc[0][1] = WMMA_BF16(af0, bf1, acc[0][1]);
    acc[1][0] = WMMA_BF16(af1, bf0, acc[1][0]);
    acc[1][1] = WMMA_BF16(af1, bf1, acc[1][1]);

    if (issuer && more) __builtin_amdgcn_s_wait_tensorcnt(0);
    __syncthreads();
    cur ^= 1;
  }

#pragma unroll
  for (int i = 0; i < 2; ++i)
#pragma unroll
    for (int j = 0; j < 2; ++j) {
      int row = m0 + wm * 32 + i * 16 + rb;
      int col = n0 + wn * 32 + j * 16 + cl;
#pragma unroll
      for (int r = 0; r < 8; ++r)
        C[(size_t)(row + r) * N + col] = acc[i][j][r];
    }
}

// ---------------------------------------------------------------------------
// dt = softplus(zx[:, 8448+h] + dt_bias[h])
// ---------------------------------------------------------------------------
__global__ __launch_bounds__(64) void dt_softplus(const float* __restrict__ zx,
                                                  const float* __restrict__ dt_bias,
                                                  float* __restrict__ dtb) {
  int row = blockIdx.x, h = threadIdx.x;
  float x = zx[(size_t)row * DINPROJ + (DINNER + CONVDIM) + h] + dt_bias[h];
  float sp = (x > 20.0f) ? x : log1pf(__expf(x));
  dtb[(size_t)row * NHEADS + h] = sp;
}

// ---------------------------------------------------------------------------
// Depthwise causal conv (width 4) + bias + silu -> bf16 xBC buffer.
// ---------------------------------------------------------------------------
__global__ __launch_bounds__(256) void conv_silu(const float* __restrict__ zx,
                                                 const float* __restrict__ cw,
                                                 const float* __restrict__ cb,
                                                 __bf16* __restrict__ xbc) {
  int ch  = blockIdx.x * 256 + threadIdx.x;   // 0..4351
  int row = blockIdx.y;                       // 0..8191
  int b = row >> 12, l = row & (SEQLEN - 1);
  float acc = cb[ch];
#pragma unroll
  for (int j = 0; j < 4; ++j) {
    int ls = l - 3 + j;
    if (ls >= 0)
      acc += cw[ch * 4 + j] * zx[(size_t)(b * SEQLEN + ls) * DINPROJ + DINNER + ch];
  }
  float o = acc / (1.0f + __expf(-acc));
  xbc[(size_t)row * CONVDIM + ch] = (__bf16)o;
}

// ---------------------------------------------------------------------------
// Fused SSD chunk scan. One block per (batch, head); 8 waves.
// B/C tiles (64x128 bf16, pitch 4352) are DMA'd into LDS by the TDM.
// ---------------------------------------------------------------------------
__global__ __launch_bounds__(256) void ssd_kernel(const __bf16* __restrict__ xbc,
                                                  const float* __restrict__ dtb,
                                                  const float* __restrict__ A_log,
                                                  const float* __restrict__ D_param,
                                                  float* __restrict__ ybuf) {
  __shared__ __bf16 Sb[64 * 128];    // state mirror (bf16), persistent
  __shared__ __bf16 Bsh[64 * 128];   // B tile [l][n]
  __shared__ __bf16 CX[64 * 128];    // phase A: C tile [l][n]; phase B: Xt+Gm
  __shared__ float  cs[64];
  __shared__ float  dtc[64];
  __bf16* Xt = CX;                   // [p][l] : x*dt transposed (later W2t)
  __bf16* Gm = CX + 64 * 64;         // [l][s] : masked decayed G

  const int tid  = threadIdx.x;
  const int lane = tid & 31;
  const int wave = tid >> 5;
  const int b = blockIdx.x >> 6;
  const int h = blockIdx.x & 63;
  const float Ah = -__expf(A_log[h]);
  const float Dh = D_param[h];

  const int cl = lane & 15;
  const int rb = (lane >> 4) << 3;

  for (int i = tid; i < 64 * 128; i += 256) Sb[i] = (__bf16)0.0f;
  v8f sacc[4];
#pragma unroll
  for (int j = 0; j < 4; ++j) sacc[j] = vzero8();
  __syncthreads();

  for (int c = 0; c < SEQLEN / CHUNK; ++c) {
    const int base = b * SEQLEN + c * CHUNK;

    // 1) dt and cumsum of A*dt ; 2) TDM-load B (Bsh) and C (CX) tiles
    if (tid < 64) {
      float d = dtb[(size_t)(base + tid) * NHEADS + h];
      dtc[tid] = d;
      cs[tid] = Ah * d;
    }
    __syncthreads();
    if (wave == 0) {
      if (lane == 0) {
        float run = 0.0f;
        for (int i = 0; i < 64; ++i) { run += cs[i]; cs[i] = run; }
      }
      tdm_load_2d_bf16(lds_off(Bsh), xbc + (size_t)base * CONVDIM + DINNER,
                       128, 64, CONVDIM);
      tdm_load_2d_bf16(lds_off(CX), xbc + (size_t)base * CONVDIM + DINNER + DSTATE,
                       128, 64, CONVDIM);
      __builtin_amdgcn_s_wait_tensorcnt(0);
    }
    __syncthreads();

    // 3+4) G = C*B^T (lower tiles) and Y_off = C*S^T, sharing the C fragment
    v8f gacc[2], yacc[2];
#pragma unroll
    for (int q = 0; q < 2; ++q) {
      int t  = wave + q * 8;
      int tm = t >> 2, tn = t & 3;
      v8f g = vzero8(), y = vzero8();
      bool gv = (tn <= tm);
#pragma unroll
      for (int kk = 0; kk < 128; kk += 32) {
        v16bf af = frag_row(CX, tm * 16, 128, kk, lane);
        if (gv) {
          v16bf bb = frag_row(Bsh, tn * 16, 128, kk, lane);
          g = WMMA_BF16(af, bb, g);
        }
        v16bf bs = frag_row(Sb, tn * 16, 128, kk, lane);
        y = WMMA_BF16(af, bs, y);
      }
      gacc[q] = g; yacc[q] = y;
    }

    // 5) Y_off row scale: exp(cs[l])
#pragma unroll
    for (int q = 0; q < 2; ++q) {
      int tm = (wave + q * 8) >> 2;
#pragma unroll
      for (int r = 0; r < 8; ++r) yacc[q][r] *= __expf(cs[tm * 16 + rb + r]);
    }
    __syncthreads();   // done reading CX (C tile) and Sb

    // 6) rebuild CX region: Xt[p][l] = x*dt (transposed) and Gm[l][s]
    {
      int p = tid >> 2, ls = (tid & 3) * 16;
#pragma unroll
      for (int i = 0; i < 16; ++i) {
        int l = ls + i;
        float xv = (float)xbc[(size_t)(base + l) * CONVDIM + h * HEADDIM + p];
        Xt[p * 64 + l] = (__bf16)(xv * dtc[l]);
      }
    }
#pragma unroll
    for (int q = 0; q < 2; ++q) {
      int t = wave + q * 8;
      int tm = t >> 2, tn = t & 3;
#pragma unroll
      for (int r = 0; r < 8; ++r) {
        int l = tm * 16 + rb + r, s = tn * 16 + cl;
        float v = 0.0f;
        if (tn <= tm && s <= l) v = gacc[q][r] * __expf(cs[l] - cs[s]);
        Gm[l * 64 + s] = (__bf16)v;
      }
    }
    __syncthreads();

    // 7) Y_diag accumulate: yacc += Gm * Xt^T
#pragma unroll
    for (int q = 0; q < 2; ++q) {
      int t = wave + q * 8;
      int tm = t >> 2, tp = t & 3;
      v8f y = yacc[q];
#pragma unroll
      for (int kk = 0; kk < 64; kk += 32) {
        v16bf af = frag_row(Gm, tm * 16, 64, kk, lane);
        v16bf bx = frag_row(Xt, tp * 16, 64, kk, lane);
        y = WMMA_BF16(af, bx, y);
      }
      yacc[q] = y;
    }

    // 8) write Y + D*x
#pragma unroll
    for (int q = 0; q < 2; ++q) {
      int t = wave + q * 8;
      int tm = t >> 2, tp = t & 3;
#pragma unroll
      for (int r = 0; r < 8; ++r) {
        int l = tm * 16 + rb + r, p = tp * 16 + cl;
        float xv = (float)xbc[(size_t)(base + l) * CONVDIM + h * HEADDIM + p];
        ybuf[(size_t)(base + l) * DINNER + h * HEADDIM + p] = yacc[q][r] + Dh * xv;
      }
    }
    __syncthreads();   // done reading Xt/Gm

    // 9) Xt -> W2t in place: *= exp(cs63 - cs[l])
    {
      float cs63 = cs[63];
      int p = tid >> 2, ls = (tid & 3) * 16;
#pragma unroll
      for (int i = 0; i < 16; ++i) {
        int l = ls + i;
        float v = (float)Xt[p * 64 + l];
        Xt[p * 64 + l] = (__bf16)(v * __expf(cs63 - cs[l]));
      }
    }
    __syncthreads();

    // 10+11) state update: S = exp(cs63)*S + W2t * B, dump bf16 mirror
    {
      float cdec = __expf(cs[63]);
#pragma unroll
      for (int j = 0; j < 4; ++j) {
        int t  = wave + j * 8;            // 0..31 state tiles (4p x 8n)
        int tp = t >> 3, tn2 = t & 7;
        v8f s = sacc[j];
#pragma unroll
        for (int r = 0; r < 8; ++r) s[r] *= cdec;
#pragma unroll
        for (int kk = 0; kk < 64; kk += 32) {
          v16bf af = frag_row(Xt, tp * 16, 64, kk, lane);          // W2t[p][l]
          v16bf bb = frag_gatherT(Bsh, tn2 * 16, 128, kk, lane);   // B[l][n] -> N-major
          s = WMMA_BF16(af, bb, s);
        }
        sacc[j] = s;
#pragma unroll
        for (int r = 0; r < 8; ++r) {
          int p = tp * 16 + rb + r, n = tn2 * 16 + cl;
          Sb[p * 128 + n] = (__bf16)s[r];
        }
      }
    }
    __syncthreads();   // Sb / Bsh / CX safe for next chunk
  }
}

// ---------------------------------------------------------------------------
// y = y * silu(z); RMSNorm over D_INNER; * norm_w; -> bf16 for out_proj
// ---------------------------------------------------------------------------
__global__ __launch_bounds__(256) void gate_rms(const float* __restrict__ ybuf,
                                                const float* __restrict__ zx,
                                                const float* __restrict__ norm_w,
                                                __bf16* __restrict__ ybf) {
  int row = blockIdx.x;
  const float* y = ybuf + (size_t)row * DINNER;
  const float* z = zx + (size_t)row * DINPROJ;
  float g[16];
  float ss = 0.0f;
#pragma unroll
  for (int j = 0; j < 16; ++j) {
    int d = threadIdx.x + j * 256;
    float zv = z[d];
    float gv = y[d] * (zv / (1.0f + __expf(-zv)));
    g[j] = gv;
    ss += gv * gv;
  }
  __shared__ float red[256];
  red[threadIdx.x] = ss;
  __syncthreads();
  for (int s = 128; s > 0; s >>= 1) {
    if (threadIdx.x < s) red[threadIdx.x] += red[threadIdx.x + s];
    __syncthreads();
  }
  float scale = rsqrtf(red[0] / (float)DINNER + 1e-5f);
#pragma unroll
  for (int j = 0; j < 16; ++j) {
    int d = threadIdx.x + j * 256;
    ybf[(size_t)row * DINNER + d] = (__bf16)(g[j] * scale * norm_w[d]);
  }
}

// ---------------------------------------------------------------------------
// Workspace layout (bytes)
// ---------------------------------------------------------------------------
static constexpr size_t OFF_UBF    = 0;                    // 8192*2048 bf16 (reused as YBF)
static constexpr size_t OFF_WINBF  = 33554432;             // 8512*2048 bf16
static constexpr size_t OFF_WOUTBF = 68419584;             // 2048*4096 bf16
static constexpr size_t OFF_ZX     = 85196800;             // 8192*8512 f32
static constexpr size_t OFF_DT     = 364118016;            // 8192*64  f32
static constexpr size_t OFF_XBC    = 366215168;            // 8192*4352 bf16
static constexpr size_t OFF_Y      = 437518336;            // 8192*4096 f32
static constexpr size_t OFF_YBF    = OFF_UBF;              // u_bf dead by then

extern "C" void kernel_launch(void* const* d_in, const int* in_sizes, int n_in,
                              void* d_out, int out_size, void* d_ws, size_t ws_size,
                              hipStream_t stream) {
  const float* u       = (const float*)d_in[0];
  const float* W_in    = (const float*)d_in[1];
  const float* conv_w  = (const float*)d_in[2];
  const float* conv_b  = (const float*)d_in[3];
  const float* dt_bias = (const float*)d_in[4];
  const float* A_log   = (const float*)d_in[5];
  const float* D_param = (const float*)d_in[6];
  const float* norm_w  = (const float*)d_in[7];
  const float* W_out   = (const float*)d_in[8];

  char* ws = (char*)d_ws;
  __bf16* u_bf    = (__bf16*)(ws + OFF_UBF);
  __bf16* win_bf  = (__bf16*)(ws + OFF_WINBF);
  __bf16* wout_bf = (__bf16*)(ws + OFF_WOUTBF);
  float*  zx      = (float*)(ws + OFF_ZX);
  float*  dtb     = (float*)(ws + OFF_DT);
  __bf16* xbc     = (__bf16*)(ws + OFF_XBC);
  float*  ybuf    = (float*)(ws + OFF_Y);
  __bf16* ybf     = (__bf16*)(ws + OFF_YBF);

  // 1) converts
  {
    int n0 = MROWS * DMODEL;        // u
    int n1 = DINPROJ * DMODEL;      // W_in
    int n2 = DMODEL * DINNER;       // W_out
    cvt_bf16<<<(n0 + 255) / 256, 256, 0, stream>>>(u, u_bf, n0);
    cvt_bf16<<<(n1 + 255) / 256, 256, 0, stream>>>(W_in, win_bf, n1);
    cvt_bf16<<<(n2 + 255) / 256, 256, 0, stream>>>(W_out, wout_bf, n2);
  }

  // 2) in_proj GEMM: zx[8192,8512] = u_bf[8192,2048] * W_in^T
  gemm_bf16_nt<<<dim3(DINPROJ / 64, MROWS / 128), 256, 0, stream>>>(
      u_bf, win_bf, zx, MROWS, DINPROJ, DMODEL);

  // 3) dt softplus, conv+silu
  dt_softplus<<<MROWS, 64, 0, stream>>>(zx, dt_bias, dtb);
  conv_silu<<<dim3(CONVDIM / 256, MROWS), 256, 0, stream>>>(zx, conv_w, conv_b, xbc);

  // 4) fused SSD scan
  ssd_kernel<<<BATCH * NHEADS, 256, 0, stream>>>(xbc, dtb, A_log, D_param, ybuf);

  // 5) gate + RMSNorm -> bf16
  gate_rms<<<MROWS, 256, 0, stream>>>(ybuf, zx, norm_w, ybf);

  // 6) out_proj GEMM: out[8192,2048] = ybf[8192,4096] * W_out^T
  gemm_bf16_nt<<<dim3(DMODEL / 64, MROWS / 128), 256, 0, stream>>>(
      ybf, wout_bf, (float*)d_out, MROWS, DMODEL, DINNER);
}
